// GATv2Conv_76570676953475
// MI455X (gfx1250) — matrix operand order repaired and checked by reading, not compiled, not run on hardware
//
#include <hip/hip_runtime.h>
#include <hip/hip_fp16.h>

typedef __attribute__((ext_vector_type(16))) _Float16 v16h;
typedef __attribute__((ext_vector_type(8)))  _Float16 v8h;
typedef __attribute__((ext_vector_type(8)))  float    v8f;
typedef __attribute__((ext_vector_type(4)))  float    v4f;

#define NN      50000      // nodes
#define NE      400000     // edges
#define DIM     256        // D_IN == EMBED
#define HEADS   8
#define HD      32         // head dim

// ---------- monotonic float <-> uint mapping for atomicMax on signed floats ----
__device__ __forceinline__ unsigned f2mono(float f) {
    unsigned u = __float_as_uint(f);
    return (u & 0x80000000u) ? ~u : (u | 0x80000000u);
}
__device__ __forceinline__ float mono2f(unsigned u) {
    return (u & 0x80000000u) ? __uint_as_float(u & 0x7FFFFFFFu)
                             : __uint_as_float(~u);
}

// mish(x) = x * tanh(softplus(x)) = x * (u^2-1)/(u^2+1), u = 1+e^x.
// Single hardware v_exp_f32; clamp arg at 20 so u^2 can't overflow to inf
// (tanh is 1 to <1e-17 there, so the clamp is exact in f32).
__device__ __forceinline__ float mish_f(float x) {
    float e  = __expf(fminf(x, 20.0f));
    float u  = 1.0f + e;
    float u2 = u * u;
    return x * ((u2 - 1.0f) / (u2 + 1.0f));
}

// ---------------------------- utility kernels ---------------------------------
__global__ void zero_f32_k(float* p, int n) {
    int i = blockIdx.x * 256 + threadIdx.x;
    if (i < n) p[i] = 0.0f;
}
__global__ void zero_u32_k(unsigned* p, int n) {
    int i = blockIdx.x * 256 + threadIdx.x;
    if (i < n) p[i] = 0u;
}

// nodes f32 -> f16 (same row-major layout)
__global__ void cvt_nodes_k(const float* __restrict__ nodes, _Float16* __restrict__ nf16) {
    int i = blockIdx.x * 256 + threadIdx.x;
    if (i < NN * DIM) nf16[i] = (_Float16)nodes[i];
}

// Pack Wi|Wj [K=256][N=256] row-major f32 into f16 B-fragments.
// Fragment layout (contiguous k order used here, 16-bit B 32x16): lane 0-15 = col n, K 0..15;
// lane 16-31 = col n, K 16..31; 16 halves contiguous per lane.
// half index = (((w*16 + nt)*8 + kt)*32 + lane)*16 + h
// maps to W[(kt*32 + (lane>=16?16:0) + h)*256 + nt*16 + (lane&15)]
__global__ void pack_w_k(const float* __restrict__ Wi, const float* __restrict__ Wj,
                         _Float16* __restrict__ packW) {
    int idx = blockIdx.x * 256 + threadIdx.x;        // < 2*16*8*32*16 = 131072
    int h    = idx & 15;
    int lane = (idx >> 4) & 31;
    int kt   = (idx >> 9) & 7;
    int nt   = (idx >> 12) & 15;
    int w    = (idx >> 16) & 1;
    int col  = (nt << 4) + (lane & 15);
    int K    = (kt << 5) + ((lane >> 4) << 4) + h;
    const float* W = w ? Wj : Wi;
    packW[idx] = (_Float16)W[K * DIM + col];
}

// ------------------------- WMMA projection GEMM -------------------------------
// Pi = nodes @ Wi + bi ; Pj = nodes @ Wj + bj.  One 16x16 tile per wave.
// tiles = 3125 (mt) * 2 (w) * 16 (nt) = 100000; 8 waves/block -> 12500 blocks.
// Consecutive waves of a block share the same M-tile (A fragments hit cache).
__global__ void gemm_proj_k(const _Float16* __restrict__ nf16,
                            const _Float16* __restrict__ packW,
                            const float* __restrict__ bi, const float* __restrict__ bj,
                            float* __restrict__ Pi, float* __restrict__ Pj) {
    const int wave = (blockIdx.x << 3) | (threadIdx.x >> 5);
    const int lane = threadIdx.x & 31;
    const int nt   = wave & 15;
    const int wsel = (wave >> 4) & 1;
    const int mt   = wave >> 5;

    // A fragment (16-bit A 16x32): lane 0-15 row m, halves = K{0..15};
    // lane 16-31 row m, halves = K{16..31}
    const int row  = (mt << 4) + (lane & 15);
    const int hi8  = ((lane >> 4) << 4);                 // 0 or 16
    const _Float16* arow  = nf16 + (size_t)row * DIM + hi8;
    const _Float16* bbase = packW + (size_t)(((wsel << 4) | nt) << 12) + (lane << 4);

    v8f c = {};
#pragma unroll
    for (int kt = 0; kt < 8; ++kt) {
        const _Float16* ap = arow + (kt << 5);
        v8h alo = *(const v8h*)(ap);                      // K base + 0..7
        v8h ahi = *(const v8h*)(ap + 8);                  // K base + 8..15
        v16h a = __builtin_shufflevector(alo, ahi,
                    0,1,2,3,4,5,6,7,8,9,10,11,12,13,14,15);
        v16h b = *(const v16h*)(bbase + (kt << 9));       // 32B contiguous
        c = __builtin_amdgcn_wmma_f32_16x16x32_f16(
                false, a, false, b, (short)0, c, false, false);
    }

    // C/D layout: lane 0-15 -> rows 0..7 in vgpr 0..7; lane 16-31 -> rows 8..15
    const int col = (nt << 4) + (lane & 15);
    const float bv = (wsel ? bj : bi)[col];
    float* Pd = wsel ? Pj : Pi;
    const int rbase = (mt << 4) + ((lane >> 4) << 3);
#pragma unroll
    for (int r = 0; r < 8; ++r)
        Pd[(size_t)(rbase + r) * DIM + col] = c[r] + bv;
}

// ---------------------- edge logits + segment max -----------------------------
// one thread per (edge, head): x = mish(Pi[s]+Pj[r]); logit = dot(x, a_w)+a_b
__global__ void edge_logits_k(const int* __restrict__ senders,
                              const int* __restrict__ receivers,
                              const float* __restrict__ Pi, const float* __restrict__ Pj,
                              const float* __restrict__ a_w, const float* __restrict__ a_b,
                              float* __restrict__ logits, unsigned* __restrict__ nmax) {
    int t = blockIdx.x * 256 + threadIdx.x;
    if (t >= NE * HEADS) return;
    int e = t >> 3, h = t & 7;
    int s = senders[e], r = receivers[e];
    const float* pi = Pi + (size_t)s * DIM + h * HD;
    const float* pj = Pj + (size_t)r * DIM + h * HD;
    float acc = 0.0f;
#pragma unroll
    for (int k = 0; k < HD; k += 4) {
        v4f a = *(const v4f*)(pi + k);
        v4f b = *(const v4f*)(pj + k);
#pragma unroll
        for (int q = 0; q < 4; ++q) {
            float x = a[q] + b[q];
            acc += mish_f(x) * a_w[k + q];
        }
    }
    float lg = acc + a_b[0];
    logits[t] = lg;
    atomicMax(&nmax[r * HEADS + h], f2mono(lg));
}

// -------------------------- softmax denominator -------------------------------
__global__ void edge_denom_k(const int* __restrict__ receivers,
                             const float* __restrict__ logits,
                             const unsigned* __restrict__ nmax,
                             float* __restrict__ denom) {
    int t = blockIdx.x * 256 + threadIdx.x;
    if (t >= NE * HEADS) return;
    int e = t >> 3, h = t & 7;
    int r = receivers[e];
    float mx = mono2f(nmax[r * HEADS + h]);
    atomicAdd(&denom[r * HEADS + h], __expf(logits[t] - mx));
}

// ------------------------- weighted aggregation -------------------------------
__global__ void aggregate_k(const int* __restrict__ senders,
                            const int* __restrict__ receivers,
                            const float* __restrict__ logits,
                            const unsigned* __restrict__ nmax,
                            const float* __restrict__ denom,
                            const float* __restrict__ Pi,
                            float* __restrict__ out) {
    int t = blockIdx.x * 256 + threadIdx.x;
    if (t >= NE * HEADS) return;
    int e = t >> 3, h = t & 7;
    int s = senders[e], r = receivers[e];
    int rh = r * HEADS + h;
    float w = __expf(logits[t] - mono2f(nmax[rh])) / denom[rh];
    const float* pi = Pi + (size_t)s * DIM + h * HD;
    float* o = out + (size_t)r * DIM + h * HD;
#pragma unroll
    for (int k = 0; k < HD; k += 4) {
        v4f a = *(const v4f*)(pi + k);
        atomicAdd(o + k + 0, a[0] * w);
        atomicAdd(o + k + 1, a[1] * w);
        atomicAdd(o + k + 2, a[2] * w);
        atomicAdd(o + k + 3, a[3] * w);
    }
}

// ------------------------------- launcher -------------------------------------
extern "C" void kernel_launch(void* const* d_in, const int* in_sizes, int n_in,
                              void* d_out, int out_size, void* d_ws, size_t ws_size,
                              hipStream_t stream) {
    const float* nodes     = (const float*)d_in[0];
    const int*   senders   = (const int*)  d_in[1];
    const int*   receivers = (const int*)  d_in[2];
    const float* Wi        = (const float*)d_in[3];
    const float* bi        = (const float*)d_in[4];
    const float* Wj        = (const float*)d_in[5];
    const float* bj        = (const float*)d_in[6];
    const float* a_w       = (const float*)d_in[7];
    const float* a_b       = (const float*)d_in[8];
    float*       out       = (float*)d_out;

    // workspace layout (all region sizes multiples of 256B)
    char* w = (char*)d_ws;
    _Float16* packW = (_Float16*)w;  w += 262144;                    // 2*256*256 f16
    _Float16* nf16  = (_Float16*)w;  w += (size_t)NN * DIM * 2;      // 25.6 MB
    float*    Pi    = (float*)w;     w += (size_t)NN * DIM * 4;      // 51.2 MB
    float*    Pj    = (float*)w;     w += (size_t)NN * DIM * 4;      // 51.2 MB
    float*    logit = (float*)w;     w += (size_t)NE * HEADS * 4;    // 12.8 MB
    unsigned* nmax  = (unsigned*)w;  w += (size_t)NN * HEADS * 4;    // 1.6 MB
    float*    denom = (float*)w;                                     // 1.6 MB

    // 1) init accumulators (d_out / scratch are poisoned)
    zero_f32_k<<<(NN * DIM + 255) / 256, 256, 0, stream>>>(out, NN * DIM);
    zero_u32_k<<<(NN * HEADS + 255) / 256, 256, 0, stream>>>(nmax, NN * HEADS);
    zero_f32_k<<<(NN * HEADS + 255) / 256, 256, 0, stream>>>(denom, NN * HEADS);

    // 2) prep: weight fragment packing + node f16 conversion
    pack_w_k<<<131072 / 256, 256, 0, stream>>>(Wi, Wj, packW);
    cvt_nodes_k<<<(NN * DIM + 255) / 256, 256, 0, stream>>>(nodes, nf16);

    // 3) WMMA projection GEMMs: 3125 M-tiles * 2 W * 16 N-tiles = 100000 tiles, 8 waves/block
    gemm_proj_k<<<12500, 256, 0, stream>>>(nf16, packW, bi, bj, Pi, Pj);

    // 4) edge logits + segment max
    edge_logits_k<<<(NE * HEADS + 255) / 256, 256, 0, stream>>>(
        senders, receivers, Pi, Pj, a_w, a_b, logit, nmax);

    // 5) softmax denominators
    edge_denom_k<<<(NE * HEADS + 255) / 256, 256, 0, stream>>>(
        receivers, logit, nmax, denom);

    // 6) weighted aggregation into out
    aggregate_k<<<(NE * HEADS + 255) / 256, 256, 0, stream>>>(
        senders, receivers, logit, nmax, denom, Pi, out);
}
